// TI_TransformerWithEdges_23244363006447
// MI455X (gfx1250) — matrix operand order repaired and checked by reading, not compiled, not run on hardware
//
#include <hip/hip_runtime.h>
#include <hip/hip_bf16.h>

#define NNODE 4096
#define NEDGE 65536
#define NEXT  (NEDGE + NNODE)
#define DH    4
#define DC    64
#define DD    256
#define NGR   16
#define ORD_NEG_INF 0x007fffffu

typedef __attribute__((ext_vector_type(16))) __bf16 v16bf;
typedef __attribute__((ext_vector_type(8)))  __bf16 v8bf;
typedef __attribute__((ext_vector_type(4)))  __bf16 v4bf;
typedef __attribute__((ext_vector_type(8)))  float  v8f;

__device__ __forceinline__ __bf16 f2bf(float f) { return (__bf16)f; }

__device__ __forceinline__ unsigned f2ord(float f) {
  unsigned u = __float_as_uint(f);
  return (u & 0x80000000u) ? ~u : (u | 0x80000000u);
}
__device__ __forceinline__ float ord2f(unsigned u) {
  unsigned v = (u & 0x80000000u) ? (u & 0x7fffffffu) : ~u;
  return __uint_as_float(v);
}

// ============ bf16 GEMM with async global->LDS DMA staging (hot path) ===============
// C[M,N] = A[M,K](bf16,row-major) * Bt[N,K](bf16, = B transposed) + bias
// Requires M%64==0, N%64==0, K%32==0. Each lane's 16B fragment slot is DMA'd straight
// into LDS in WMMA fragment order: no VGPR staging, no ds_store, no converts.
__global__ __launch_bounds__(128) void k_gemm_bfa(
    const __bf16* __restrict__ A, const __bf16* __restrict__ Bt,
    const float* __restrict__ bias, float* __restrict__ C,
    int M, int N, int K)
{
  __shared__ v16bf sAf[4][32];   // [wave(m-subtile)][lane]
  __shared__ v16bf sBf[4][32];   // [n-subtile t][lane]
  const int tid  = threadIdx.x;
  const int wave = tid >> 5, lane = tid & 31;
  const int half = lane >> 4, lo = lane & 15;
  const int bm = blockIdx.x * 64, bn = blockIdx.y * 64;
  v8f acc[4] = {};

#pragma unroll 1
  for (int kb = 0; kb < K; kb += 32) {
    // A tile: per-lane 16B DMA to frag-order LDS address (ASYNCcnt)
#pragma unroll
    for (int it = 0; it < 2; ++it) {
      int s = tid + it * 128;
      int r = s >> 2, kg = s & 3;
      const __bf16* gp = &A[(size_t)(bm + r) * K + kb + kg * 8];
      unsigned loff = (unsigned)(uintptr_t)
          &((v8bf*)&sAf[r >> 4][(kg & 1) * 16 + (r & 15)])[kg >> 1];
      asm volatile("global_load_async_to_lds_b128 %0, %1, off"
                   :: "v"(loff), "v"(gp) : "memory");
    }
    // B tile from transposed weights: contiguous 16B per frag slot
#pragma unroll
    for (int it = 0; it < 2; ++it) {
      int s = tid + it * 128;
      int n = s >> 2, kg = s & 3;
      const __bf16* gp = &Bt[(size_t)(bn + n) * K + kb + kg * 8];
      unsigned loff = (unsigned)(uintptr_t)
          &((v8bf*)&sBf[n >> 4][(kg & 1) * 16 + (n & 15)])[kg >> 1];
      asm volatile("global_load_async_to_lds_b128 %0, %1, off"
                   :: "v"(loff), "v"(gp) : "memory");
    }
    asm volatile("s_wait_asynccnt 0x0" ::: "memory");
    __syncthreads();

    v16bf a = sAf[wave][lane];
#pragma unroll
    for (int t = 0; t < 4; ++t) {
      v16bf b = sBf[t][lane];
      acc[t] = __builtin_amdgcn_wmma_f32_16x16x32_bf16(
          false, a, false, b, (short)0, acc[t], false, false);
    }
    __syncthreads();
  }

#pragma unroll
  for (int t = 0; t < 4; ++t) {
    int gn = bn + t * 16 + lo;
    float bv = bias ? bias[gn] : 0.f;
#pragma unroll
    for (int j = 0; j < 8; ++j) {
      int gm = bm + wave * 16 + half * 8 + j;
      C[(size_t)gm * N + gn] = acc[t][j] + bv;
    }
  }
}

// ============ f32-input GEMM (cold shapes: feat, GAT layer0, lin1/lin2) =============
__global__ __launch_bounds__(128) void k_gemm(
    const float* __restrict__ A, const float* __restrict__ B,
    const float* __restrict__ bias, float* __restrict__ C,
    int M, int N, int K, int relu)
{
  __shared__ v16bf sAf[4][32];
  __shared__ v16bf sBf[4][32];
  const int tid  = threadIdx.x;
  const int wave = tid >> 5, lane = tid & 31;
  const int half = lane >> 4, lo = lane & 15;
  const int bm = blockIdx.x * 64, bn = blockIdx.y * 64;
  v8f acc[4] = {};

#pragma unroll 1
  for (int kb = 0; kb < K; kb += 32) {
#pragma unroll
    for (int it = 0; it < 2; ++it) {
      int s = tid + it * 128;
      int r = s >> 2, kg = s & 3;
      int gr = bm + r;
      int cr = gr < M ? gr : M - 1;
      v8bf pk;
#pragma unroll
      for (int j = 0; j < 8; ++j) {
        int gk = kb + kg * 8 + j;
        int ck = gk < K ? gk : K - 1;
        float v = A[(size_t)cr * K + ck];            // always in-range address
        pk[j] = f2bf((gr < M && gk < K) ? v : 0.f);  // branchless select
      }
      ((v8bf*)&sAf[r >> 4][(kg & 1) * 16 + (r & 15)])[kg >> 1] = pk;
    }
#pragma unroll
    for (int it = 0; it < 2; ++it) {
      int s = tid + it * 128;
      int kg = s >> 6, n = s & 63;
      int gn = bn + n;
      int cn = gn < N ? gn : N - 1;
      v8bf pk;
#pragma unroll
      for (int j = 0; j < 8; ++j) {
        int gk = kb + kg * 8 + j;
        int ck = gk < K ? gk : K - 1;
        float v = B[(size_t)ck * N + cn];
        pk[j] = f2bf((gk < K && gn < N) ? v : 0.f);
      }
      ((v8bf*)&sBf[n >> 4][(kg & 1) * 16 + (n & 15)])[kg >> 1] = pk;
    }
    __syncthreads();
    v16bf a = sAf[wave][lane];
#pragma unroll
    for (int t = 0; t < 4; ++t) {
      v16bf b = sBf[t][lane];
      acc[t] = __builtin_amdgcn_wmma_f32_16x16x32_bf16(
          false, a, false, b, (short)0, acc[t], false, false);
    }
    __syncthreads();
  }

#pragma unroll
  for (int t = 0; t < 4; ++t)
#pragma unroll
    for (int j = 0; j < 8; ++j) {
      int gm = bm + wave * 16 + half * 8 + j;
      int gn = bn + t * 16 + lo;
      if (gm < M && gn < N) {
        float v = acc[t][j];
        if (bias) v += bias[gn];
        if (relu) v = fmaxf(v, 0.f);
        C[(size_t)gm * N + gn] = v;
      }
    }
}

// ---------------- Flash-style attention, one wave per (16-query tile, head) ----------
__global__ __launch_bounds__(32) void k_attention(
    const float* __restrict__ Q, const float* __restrict__ Kp,
    const float* __restrict__ V, __bf16* __restrict__ O, int nkv)
{
  __shared__ v16bf sQf[2][32];      // [c-chunk][lane]   A-frag order
  __shared__ v16bf sKf[2][2][32];   // [key-sub][c-chunk][lane] B-frag (contraction=c)
  __shared__ v16bf sVf[4][32];      // [c-sub t][lane]   B-frag (contraction=key)
  __shared__ v16bf sPf[32];         // P 16x32 A-frag order
  __bf16* sPb = (__bf16*)sPf;

  const int lane = threadIdx.x;
  const int half = lane >> 4, lo = lane & 15;
  const int h  = blockIdx.y;
  const int q0 = blockIdx.x * 16;

#pragma unroll
  for (int it = 0; it < 4; ++it) {
    int s = lane + it * 32;
    int r = s >> 3, cg = s & 7;
    const float4* qp = (const float4*)&Q[(size_t)(q0 + r) * DD + h * DC + cg * 8];
    float4 f0 = qp[0], f1 = qp[1];
    v8bf pk;
    pk[0] = f2bf(f0.x * 0.125f); pk[1] = f2bf(f0.y * 0.125f);
    pk[2] = f2bf(f0.z * 0.125f); pk[3] = f2bf(f0.w * 0.125f);
    pk[4] = f2bf(f1.x * 0.125f); pk[5] = f2bf(f1.y * 0.125f);
    pk[6] = f2bf(f1.z * 0.125f); pk[7] = f2bf(f1.w * 0.125f);
    int c2 = cg & 3;
    ((v8bf*)&sQf[cg >> 2][(c2 & 1) * 16 + r])[c2 >> 1] = pk;
  }
  __syncthreads();
  v16bf aQ0 = sQf[0][lane];
  v16bf aQ1 = sQf[1][lane];

  v8f o0 = {}, o1 = {}, o2 = {}, o3 = {};
  float rmax[8], rsum[8];
#pragma unroll
  for (int j = 0; j < 8; ++j) { rmax[j] = -__builtin_inff(); rsum[j] = 0.f; }

#pragma unroll 1
  for (int kb = 0; kb < nkv; kb += 32) {
    __syncthreads();
#pragma unroll
    for (int it = 0; it < 8; ++it) {
      int s = lane + it * 32;
      int key = s >> 3, cg = s & 7;
      const float4* kp = (const float4*)&Kp[(size_t)(kb + key) * DD + h * DC + cg * 8];
      float4 f0 = kp[0], f1 = kp[1];
      v8bf pk;
      pk[0] = f2bf(f0.x); pk[1] = f2bf(f0.y); pk[2] = f2bf(f0.z); pk[3] = f2bf(f0.w);
      pk[4] = f2bf(f1.x); pk[5] = f2bf(f1.y); pk[6] = f2bf(f1.z); pk[7] = f2bf(f1.w);
      int c2 = cg & 3;
      ((v8bf*)&sKf[key >> 4][cg >> 2][(c2 & 1) * 16 + (key & 15)])[c2 >> 1] = pk;
    }
#pragma unroll
    for (int it = 0; it < 8; ++it) {
      int s = lane + it * 32;
      int kg = s >> 6, c = s & 63;
      const float* vp = &V[(size_t)(kb + kg * 8) * DD + h * DC + c];
      v8bf pk;
#pragma unroll
      for (int j = 0; j < 8; ++j) pk[j] = f2bf(vp[(size_t)j * DD]);
      ((v8bf*)&sVf[c >> 4][(kg & 1) * 16 + (c & 15)])[kg >> 1] = pk;
    }
    __syncthreads();

    v8f s0 = {}, s1 = {};
    {
      v16bf b;
      b = sKf[0][0][lane];
      s0 = __builtin_amdgcn_wmma_f32_16x16x32_bf16(false, aQ0, false, b, (short)0, s0, false, false);
      b = sKf[0][1][lane];
      s0 = __builtin_amdgcn_wmma_f32_16x16x32_bf16(false, aQ1, false, b, (short)0, s0, false, false);
      b = sKf[1][0][lane];
      s1 = __builtin_amdgcn_wmma_f32_16x16x32_bf16(false, aQ0, false, b, (short)0, s1, false, false);
      b = sKf[1][1][lane];
      s1 = __builtin_amdgcn_wmma_f32_16x16x32_bf16(false, aQ1, false, b, (short)0, s1, false, false);
    }

#pragma unroll
    for (int j = 0; j < 8; ++j) {
      float tm = fmaxf(s0[j], s1[j]);
      tm = fmaxf(tm, __shfl_xor(tm, 1, 16));
      tm = fmaxf(tm, __shfl_xor(tm, 2, 16));
      tm = fmaxf(tm, __shfl_xor(tm, 4, 16));
      tm = fmaxf(tm, __shfl_xor(tm, 8, 16));
      float nm = fmaxf(rmax[j], tm);
      float sc = expf(rmax[j] - nm);
      float p0 = expf(s0[j] - nm);
      float p1 = expf(s1[j] - nm);
      float ps = p0 + p1;
      ps += __shfl_xor(ps, 1, 16);
      ps += __shfl_xor(ps, 2, 16);
      ps += __shfl_xor(ps, 4, 16);
      ps += __shfl_xor(ps, 8, 16);
      rsum[j] = rsum[j] * sc + ps;
      rmax[j] = nm;
      o0[j] *= sc; o1[j] *= sc; o2[j] *= sc; o3[j] *= sc;
      int m = half * 8 + j;
      int dl = ((lo >> 3) & 1) * 16 + m;
      sPb[dl * 16 + (lo & 7)]     = f2bf(p0);
      sPb[dl * 16 + (lo & 7) + 8] = f2bf(p1);
    }
    __syncthreads();

    v16bf aP = sPf[lane];
    {
      v16bf b;
      b = sVf[0][lane];
      o0 = __builtin_amdgcn_wmma_f32_16x16x32_bf16(false, aP, false, b, (short)0, o0, false, false);
      b = sVf[1][lane];
      o1 = __builtin_amdgcn_wmma_f32_16x16x32_bf16(false, aP, false, b, (short)0, o1, false, false);
      b = sVf[2][lane];
      o2 = __builtin_amdgcn_wmma_f32_16x16x32_bf16(false, aP, false, b, (short)0, o2, false, false);
      b = sVf[3][lane];
      o3 = __builtin_amdgcn_wmma_f32_16x16x32_bf16(false, aP, false, b, (short)0, o3, false, false);
    }
  }

#pragma unroll
  for (int j = 0; j < 8; ++j) {
    int node = q0 + half * 8 + j;
    float inv = 1.f / rsum[j];
    O[(size_t)node * DD + h * DC + lo]      = f2bf(o0[j] * inv);
    O[(size_t)node * DD + h * DC + 16 + lo] = f2bf(o1[j] * inv);
    O[(size_t)node * DD + h * DC + 32 + lo] = f2bf(o2[j] * inv);
    O[(size_t)node * DD + h * DC + 48 + lo] = f2bf(o3[j] * inv);
  }
}

// ---------------- small / irregular kernels -----------------------------------------
__global__ void k_fill_u32(unsigned* p, unsigned v, int n) {
  int i = blockIdx.x * blockDim.x + threadIdx.x;
  if (i < n) p[i] = v;
}

// weight convert + transpose: Wt[n*Kd+k] = bf16(W[k*Nd+n])
__global__ void k_cvt_wT(const float* __restrict__ W, __bf16* __restrict__ Wt,
                         int Kd, int Nd) {
  int i = blockIdx.x * blockDim.x + threadIdx.x;
  if (i >= Kd * Nd) return;
  int n = i / Kd, k = i - n * Kd;
  Wt[i] = f2bf(W[(size_t)k * Nd + n]);
}

__global__ void k_node_att(const float* __restrict__ xs, const float* __restrict__ as_,
                           const float* __restrict__ ad_, float* __restrict__ als,
                           float* __restrict__ ald) {
  int t = blockIdx.x * blockDim.x + threadIdx.x;
  if (t >= NNODE * DH) return;
  int n = t >> 2, h = t & 3;
  float s = 0.f, d = 0.f;
  for (int c = 0; c < DC; ++c) {
    float v = xs[(size_t)n * DD + h * DC + c];
    s += v * as_[h * DC + c];
    d += v * ad_[h * DC + c];
  }
  als[t] = s; ald[t] = d;
}

__global__ void k_we_reduce(const float* __restrict__ We, const float* __restrict__ ae,
                            float* __restrict__ M) {
  int t = threadIdx.x;
  if (t >= 8) return;
  int d = t >> 2, h = t & 3;
  float s = 0.f;
  for (int c = 0; c < DC; ++c) s += We[d * DD + h * DC + c] * ae[h * DC + c];
  M[t] = s;
}

__global__ void k_edge_cnt(const int* __restrict__ ei, const float* __restrict__ ea,
                           float* __restrict__ cnt, float* __restrict__ easum) {
  int e = blockIdx.x * blockDim.x + threadIdx.x;
  if (e >= NEDGE) return;
  int dst = ei[NEDGE + e];
  atomicAdd(&cnt[dst], 1.f);
  atomicAdd(&easum[dst * 2 + 0], ea[e * 2 + 0]);
  atomicAdd(&easum[dst * 2 + 1], ea[e * 2 + 1]);
}

__global__ void k_loop_ea(const float* __restrict__ cnt, const float* __restrict__ easum,
                          float* __restrict__ lea) {
  int n = blockIdx.x * blockDim.x + threadIdx.x;
  if (n >= NNODE) return;
  float c = fmaxf(cnt[n], 1.f);
  lea[n * 2 + 0] = easum[n * 2 + 0] / c;
  lea[n * 2 + 1] = easum[n * 2 + 1] / c;
}

__global__ void k_edge_logits(const int* __restrict__ ei, const float* __restrict__ ea,
                              const float* __restrict__ lea, const float* __restrict__ als,
                              const float* __restrict__ ald, const float* __restrict__ M,
                              float* __restrict__ al, unsigned* __restrict__ mrow) {
  int e = blockIdx.x * blockDim.x + threadIdx.x;
  if (e >= NEXT) return;
  int src, dst; float e0, e1;
  if (e < NEDGE) { src = ei[e]; dst = ei[NEDGE + e]; e0 = ea[e * 2]; e1 = ea[e * 2 + 1]; }
  else { src = dst = e - NEDGE; e0 = lea[src * 2]; e1 = lea[src * 2 + 1]; }
#pragma unroll
  for (int h = 0; h < DH; ++h) {
    float v = als[src * DH + h] + ald[dst * DH + h] + e0 * M[h] + e1 * M[DH + h];
    v = (v > 0.f) ? v : 0.2f * v;                   // leaky_relu 0.2
    al[(size_t)e * DH + h] = v;
    atomicMax(&mrow[dst * DH + h], f2ord(v));
  }
}

__global__ void k_edge_exp(const int* __restrict__ ei, const float* __restrict__ al,
                           const unsigned* __restrict__ mrow, float* __restrict__ ex,
                           float* __restrict__ den) {
  int e = blockIdx.x * blockDim.x + threadIdx.x;
  if (e >= NEXT) return;
  int dst = (e < NEDGE) ? ei[NEDGE + e] : (e - NEDGE);
#pragma unroll
  for (int h = 0; h < DH; ++h) {
    float v = expf(al[(size_t)e * DH + h] - ord2f(mrow[dst * DH + h]));
    ex[(size_t)e * DH + h] = v;
    atomicAdd(&den[dst * DH + h], v);
  }
}

__global__ __launch_bounds__(256) void k_edge_agg(
    const int* __restrict__ ei, const float* __restrict__ ex,
    const float* __restrict__ den, const float* __restrict__ xs,
    float* __restrict__ aggr) {
  int e = blockIdx.x;
  int t = threadIdx.x;
  int src, dst;
  if (e < NEDGE) { src = ei[e]; dst = ei[NEDGE + e]; }
  else { src = dst = e - NEDGE; }
  int h = t >> 6;
  float alpha = ex[(size_t)e * DH + h] / (den[dst * DH + h] + 1e-16f);
  atomicAdd(&aggr[(size_t)dst * DD + t], alpha * xs[(size_t)src * DD + t]);
}

// bias + relu, dual f32/bf16 output (bf16 feeds async-DMA GEMMs)
__global__ void k_bias_relu(const float* __restrict__ a, const float* __restrict__ b,
                            float* __restrict__ o, __bf16* __restrict__ obf) {
  int i = (blockIdx.x * blockDim.x + threadIdx.x) * 4;
  if (i >= NNODE * DD) return;
  float4 v = *(const float4*)&a[i];
  int c = i & (DD - 1);
  v.x = fmaxf(v.x + b[c + 0], 0.f);
  v.y = fmaxf(v.y + b[c + 1], 0.f);
  v.z = fmaxf(v.z + b[c + 2], 0.f);
  v.w = fmaxf(v.w + b[c + 3], 0.f);
  *(float4*)&o[i] = v;
  v4bf pb; pb[0] = f2bf(v.x); pb[1] = f2bf(v.y); pb[2] = f2bf(v.z); pb[3] = f2bf(v.w);
  *(v4bf*)&obf[i] = pb;
}

__global__ __launch_bounds__(256) void k_add_ln(
    const float* __restrict__ a, const float* __restrict__ b,
    const float* __restrict__ g, const float* __restrict__ beta,
    float* __restrict__ out, __bf16* __restrict__ obf) {
  __shared__ float red[256];
  int row = blockIdx.x, t = threadIdx.x;
  float v = a[(size_t)row * DD + t] + b[(size_t)row * DD + t];
  red[t] = v; __syncthreads();
  for (int s = 128; s > 0; s >>= 1) { if (t < s) red[t] += red[t + s]; __syncthreads(); }
  float mean = red[0] * (1.f / DD);
  __syncthreads();
  float d = v - mean;
  red[t] = d * d; __syncthreads();
  for (int s = 128; s > 0; s >>= 1) { if (t < s) red[t] += red[t + s]; __syncthreads(); }
  float var = red[0] * (1.f / DD);
  float r = d * rsqrtf(var + 1e-5f) * g[t] + beta[t];
  out[(size_t)row * DD + t] = r;
  obf[(size_t)row * DD + t] = f2bf(r);
}

__global__ void k_pool_max(const float* __restrict__ t, const int* __restrict__ batch,
                           unsigned* __restrict__ pooled) {
  int i = blockIdx.x * blockDim.x + threadIdx.x;
  if (i >= NNODE * DD) return;
  int n = i >> 8, c = i & 255;
  atomicMax(&pooled[batch[n] * DD + c], f2ord(t[i]));
}

__global__ __launch_bounds__(128) void k_head(
    const unsigned* __restrict__ pooled, const float* __restrict__ w1,
    const float* __restrict__ b1, const float* __restrict__ w2,
    const float* __restrict__ b2, float* __restrict__ out) {
  __shared__ float gg[256];
  __shared__ float s1[128];
  int g = blockIdx.x, t = threadIdx.x;
  gg[t]       = ord2f(pooled[g * DD + t]);
  gg[t + 128] = ord2f(pooled[g * DD + t + 128]);
  __syncthreads();
  float acc = b1[t];
  for (int c = 0; c < 256; ++c) acc += gg[c] * w1[c * 128 + t];
  s1[t] = fmaxf(acc, 0.f);
  __syncthreads();
  if (t < 10) {
    float o = b2[t];
    for (int c = 0; c < 128; ++c) o += s1[c] * w2[c * 10 + t];
    out[g * 10 + t] = o;
  }
}

// ---------------- orchestration -----------------------------------------------------
extern "C" void kernel_launch(void* const* d_in, const int* in_sizes, int n_in,
                              void* d_out, int out_size, void* d_ws, size_t ws_size,
                              hipStream_t stream) {
  (void)in_sizes; (void)n_in; (void)out_size; (void)ws_size;
  const float* x     = (const float*)d_in[0];
  const int*   ei    = (const int*)d_in[1];
  const float* ea    = (const float*)d_in[2];
  const int*   batch = (const int*)d_in[3];
  const float* feat_w = (const float*)d_in[4];
  const float* feat_b = (const float*)d_in[5];
  const float* saW[4] = {(const float*)d_in[30], (const float*)d_in[31],
                         (const float*)d_in[32], (const float*)d_in[33]};
  const float* saB[4] = {(const float*)d_in[34], (const float*)d_in[35],
                         (const float*)d_in[36], (const float*)d_in[37]};
  const float* caW[4] = {(const float*)d_in[38], (const float*)d_in[39],
                         (const float*)d_in[40], (const float*)d_in[41]};
  const float* caB[4] = {(const float*)d_in[42], (const float*)d_in[43],
                         (const float*)d_in[44], (const float*)d_in[45]};
  const float* lin1w = (const float*)d_in[46];
  const float* lin1b = (const float*)d_in[47];
  const float* lin2w = (const float*)d_in[48];
  const float* lin2b = (const float*)d_in[49];
  const float* ln1g = (const float*)d_in[50]; const float* ln1b = (const float*)d_in[51];
  const float* ln2g = (const float*)d_in[52]; const float* ln2b = (const float*)d_in[53];
  const float* ln3g = (const float*)d_in[54]; const float* ln3b = (const float*)d_in[55];
  const float* fc1w = (const float*)d_in[56]; const float* fc1b = (const float*)d_in[57];
  const float* fc2w = (const float*)d_in[58]; const float* fc2b = (const float*)d_in[59];

  const size_t BIG = (size_t)NNODE * DD;
  float* ws   = (float*)d_ws;
  float* hA   = ws;             // GAT hidden / memory (f32)
  float* xsB  = hA + BIG;       // xs per layer (f32)
  float* agg  = xsB + BIG;      // aggregation; also Wo/lin2 projection
  float* tB   = agg + BIG;      // transformer stream (f32)
  float* qB   = tB + BIG;
  float* kB   = qB + BIG;
  float* vB   = kB + BIG;
  float* sm   = vB + BIG;
  float* als  = sm;               sm += NNODE * DH;
  float* ald  = sm;               sm += NNODE * DH;
  float* albuf = sm;              sm += (size_t)NEXT * DH;
  float* exbuf = sm;              sm += (size_t)NEXT * DH;
  float* den  = sm;               sm += NNODE * DH;
  unsigned* mrow = (unsigned*)sm; sm += NNODE * DH;
  float* cnt  = sm;               sm += NNODE;
  float* easum = sm;              sm += NNODE * 2;
  float* lea  = sm;               sm += NNODE * 2;
  float* Msm  = sm;               sm += 32;
  float* ff1  = sm;               sm += (size_t)NNODE * 24;
  unsigned* pooled = (unsigned*)sm; sm += NGR * DD;
  __bf16* hAbf  = (__bf16*)sm;    sm += BIG / 2;   // bf16 copies of activations
  __bf16* tBbf  = (__bf16*)sm;    sm += BIG / 2;
  __bf16* aObf  = (__bf16*)sm;    sm += BIG / 2;   // attention output (bf16)
  __bf16* wT    = (__bf16*)sm;                     // 11 transposed bf16 weights
  __bf16* wTg[3]  = {wT,               wT + 1 * 65536, wT + 2 * 65536};
  __bf16* wTsa[4] = {wT + 3 * 65536,  wT + 4 * 65536, wT + 5 * 65536, wT + 6 * 65536};
  __bf16* wTca[4] = {wT + 7 * 65536,  wT + 8 * 65536, wT + 9 * 65536, wT + 10 * 65536};

  auto gemm = [&](const float* A, const float* B, const float* bias, float* C,
                  int M, int N, int K, int relu) {
    dim3 grid((M + 63) / 64, (N + 63) / 64);
    k_gemm<<<grid, 128, 0, stream>>>(A, B, bias, C, M, N, K, relu);
  };
  auto gemmbf = [&](const __bf16* A, const __bf16* Bt, const float* bias, float* C) {
    dim3 grid(NNODE / 64, DD / 64);
    k_gemm_bfa<<<grid, 128, 0, stream>>>(A, Bt, bias, C, NNODE, DD, DD);
  };

  // one-time (per launch) weight convert+transpose for async-DMA GEMMs
  for (int j = 0; j < 3; ++j)
    k_cvt_wT<<<256, 256, 0, stream>>>((const float*)d_in[6 + (j + 1) * 6], wTg[j], DD, DD);
  for (int j = 0; j < 4; ++j) {
    k_cvt_wT<<<256, 256, 0, stream>>>(saW[j], wTsa[j], DD, DD);
    k_cvt_wT<<<256, 256, 0, stream>>>(caW[j], wTca[j], DD, DD);
  }

  // feature embed: h = x @ feat_w + feat_b  -> [4096,4]
  gemm(x, feat_w, feat_b, hA, NNODE, 4, 8, 0);

  // self-loop edge-attr mean (invariant across layers)
  hipMemsetAsync(cnt, 0, NNODE * sizeof(float), stream);
  hipMemsetAsync(easum, 0, NNODE * 2 * sizeof(float), stream);
  k_edge_cnt<<<(NEDGE + 255) / 256, 256, 0, stream>>>(ei, ea, cnt, easum);
  k_loop_ea<<<(NNODE + 255) / 256, 256, 0, stream>>>(cnt, easum, lea);

  for (int i = 0; i < 4; ++i) {
    const float* gWe = (const float*)d_in[6 + i * 6 + 1];
    const float* gas = (const float*)d_in[6 + i * 6 + 2];
    const float* gad = (const float*)d_in[6 + i * 6 + 3];
    const float* gae = (const float*)d_in[6 + i * 6 + 4];
    const float* gb  = (const float*)d_in[6 + i * 6 + 5];
    if (i == 0) {
      gemm(hA, (const float*)d_in[6], nullptr, xsB, NNODE, DD, 4, 0);
    } else {
      gemmbf(hAbf, wTg[i - 1], nullptr, xsB);
    }
    k_we_reduce<<<1, 8, 0, stream>>>(gWe, gae, Msm);
    k_node_att<<<(NNODE * DH + 255) / 256, 256, 0, stream>>>(xsB, gas, gad, als, ald);
    k_fill_u32<<<(NNODE * DH + 255) / 256, 256, 0, stream>>>(mrow, ORD_NEG_INF, NNODE * DH);
    hipMemsetAsync(den, 0, NNODE * DH * sizeof(float), stream);
    hipMemsetAsync(agg, 0, BIG * sizeof(float), stream);
    k_edge_logits<<<(NEXT + 255) / 256, 256, 0, stream>>>(ei, ea, lea, als, ald, Msm, albuf, mrow);
    k_edge_exp<<<(NEXT + 255) / 256, 256, 0, stream>>>(ei, albuf, mrow, exbuf, den);
    k_edge_agg<<<NEXT, 256, 0, stream>>>(ei, exbuf, den, xsB, agg);
    k_bias_relu<<<(NNODE * DD / 4 + 255) / 256, 256, 0, stream>>>(agg, gb, hA, hAbf);
  }

  // transformer decoder layer: t = hA (copy), mem = hA/hAbf (kept)
  hipMemcpyAsync(tB, hA, BIG * sizeof(float), hipMemcpyDeviceToDevice, stream);
  dim3 attnGrid(NNODE / 16, DH);

  // self-attention (t == hA here, so A operand = hAbf)
  gemmbf(hAbf, wTsa[0], saB[0], qB);
  gemmbf(hAbf, wTsa[1], saB[1], kB);
  gemmbf(hAbf, wTsa[2], saB[2], vB);
  k_attention<<<attnGrid, 32, 0, stream>>>(qB, kB, vB, aObf, NNODE);
  gemmbf(aObf, wTsa[3], saB[3], agg);
  k_add_ln<<<NNODE, 256, 0, stream>>>(tB, agg, ln1g, ln1b, tB, tBbf);

  // cross-attention vs GAT memory
  gemmbf(tBbf, wTca[0], caB[0], qB);
  gemmbf(hAbf, wTca[1], caB[1], kB);
  gemmbf(hAbf, wTca[2], caB[2], vB);
  k_attention<<<attnGrid, 32, 0, stream>>>(qB, kB, vB, aObf, NNODE);
  gemmbf(aObf, wTca[3], caB[3], agg);
  k_add_ln<<<NNODE, 256, 0, stream>>>(tB, agg, ln2g, ln2b, tB, tBbf);

  // feed-forward (odd shapes -> f32 path)
  gemm(tB, lin1w, lin1b, ff1, NNODE, 24, DD, 1);
  gemm(ff1, lin2w, lin2b, agg, NNODE, DD, 24, 0);
  k_add_ln<<<NNODE, 256, 0, stream>>>(tB, agg, ln3g, ln3b, tB, tBbf);

  // global max pool + classifier head
  k_fill_u32<<<(NGR * DD + 255) / 256, 256, 0, stream>>>(pooled, ORD_NEG_INF, NGR * DD);
  k_pool_max<<<(NNODE * DD + 255) / 256, 256, 0, stream>>>(tB, batch, pooled);
  k_head<<<NGR, 128, 0, stream>>>(pooled, fc1w, fc1b, fc2w, fc2b, (float*)d_out);
}
